// ColightNet_22617297781066
// MI455X (gfx1250) — matrix operand order repaired and checked by reading, not compiled, not run on hardware
//
#include <hip/hip_runtime.h>
#include <hip/hip_bf16.h>
#include <math.h>

typedef __attribute__((ext_vector_type(16))) _Float16 v16h;
typedef __attribute__((ext_vector_type(8)))  float    v8f;

#define NV 5
#define DVDIM 16
#define PROJ 80   // NV*DVDIM

// ---------------------------------------------------------------------------
// Load 8 consecutive f32 -> f16 fragment elements (validity is compile-time
// at every call site: blocks of 16 K-values are either fully valid or not).
// ---------------------------------------------------------------------------
__device__ __forceinline__ void loadcvt8(v16h& a, int base, const float* __restrict__ p,
                                         int k, bool valid) {
    if (valid) {
        float4 p0 = *(const float4*)(p + k);
        float4 p1 = *(const float4*)(p + k + 4);
        a[base + 0] = (_Float16)p0.x; a[base + 1] = (_Float16)p0.y;
        a[base + 2] = (_Float16)p0.z; a[base + 3] = (_Float16)p0.w;
        a[base + 4] = (_Float16)p1.x; a[base + 5] = (_Float16)p1.y;
        a[base + 6] = (_Float16)p1.z; a[base + 7] = (_Float16)p1.w;
    } else {
#pragma unroll
        for (int j = 0; j < 8; ++j) a[base + j] = (_Float16)0.0f;
    }
}

// ---------------------------------------------------------------------------
// Pack W[K,M] f32 -> f16 in WMMA B-fragment layout:
//   Bp[(((ct*KC)+c)*32 + lane)*16 + j] = W[c*32 + (j>>3)*16 + (lane>>4)*8 + (j&7), ct*16 + (lane&15)]
// ---------------------------------------------------------------------------
__global__ void pack_w_f16(const float* __restrict__ W, _Float16* __restrict__ Bp,
                           int K, int M) {
    int idx = blockIdx.x * blockDim.x + threadIdx.x;
    int KC = (K + 31) >> 5;
    int total = (M >> 4) * KC * 512;
    if (idx >= total) return;
    int j    = idx & 15;
    int lane = (idx >> 4) & 31;
    int rest = idx >> 9;            // ct*KC + c
    int c    = rest % KC;
    int ct   = rest / KC;
    int k    = c * 32 + (j >> 3) * 16 + (lane >> 4) * 8 + (j & 7);
    int col  = ct * 16 + (lane & 15);
    Bp[idx] = (_Float16)((k < K) ? W[(size_t)k * M + col] : 0.0f);
}

// ---------------------------------------------------------------------------
// C[N,M] = act(A[N,K] @ W[K,M] + bias[M]); K compile-time (multiple of 16),
// M runtime but always a multiple of 16. 8 waves/block, one 16x16 tile/wave.
// grid = ( M/16, ceil(N/128) )
// ---------------------------------------------------------------------------
template<int K, bool RELU>
__global__ void __launch_bounds__(256) wmma_gemm_kernel(
    const float* __restrict__ A, const _Float16* __restrict__ Bp,
    const float* __restrict__ bias, float* __restrict__ C,
    int N, int M)
{
    static_assert(K % 16 == 0, "K must be multiple of 16");
    constexpr int KC = (K + 31) / 32;

    const int lane = threadIdx.x & 31;
    const int wave = threadIdx.x >> 5;
    const int half = lane >> 4;
    const int lr   = lane & 15;

    const int colBase = blockIdx.x * 16;
    const int rowBase = (blockIdx.y * 8 + wave) * 16;
    if (rowBase >= N) return;                  // wave-uniform exit

    int arowIdx = rowBase + lr;
    if (arowIdx >= N) arowIdx = N - 1;         // clamp; extra rows never stored
    const float* arow = A + (size_t)arowIdx * K;
    const v16h*  bp   = (const v16h*)Bp + (size_t)blockIdx.x * KC * 32 + lane;

    v8f acc = {};
#pragma unroll
    for (int c = 0; c < KC; ++c) {
        v16h a;
        // A layout: half 0 -> K = {c*32+0..7, c*32+16..23}, half 1 -> +8
        loadcvt8(a, 0, arow, c * 32 + half * 8,      (c * 32 + 16) <= K);
        loadcvt8(a, 8, arow, c * 32 + 16 + half * 8, (c * 32 + 32) <= K);
        v16h b = bp[c * 32];                   // pre-packed, aligned 32B
        acc = __builtin_amdgcn_wmma_f32_16x16x32_f16(
            false, a, false, b, (short)0, acc, false, false);
    }

    const int col = colBase + lr;              // always < M (M % 16 == 0)
    const float bv = bias[col];
#pragma unroll
    for (int v = 0; v < 8; ++v) {
        const int orow = rowBase + v + half * 8;
        if (orow < N) {
            float val = acc[v] + bv;
            if (RELU) val = fmaxf(val, 0.0f);
            C[(size_t)orow * M + col] = val;
        }
    }
}

// ---------------------------------------------------------------------------
// Segment-softmax edge kernels
// ---------------------------------------------------------------------------
__device__ __forceinline__ void atomicMaxF(float* addr, float val) {
    if (val >= 0.0f) atomicMax((int*)addr, __float_as_int(val));
    else             atomicMin((unsigned int*)addr, __float_as_uint(val));
}

__global__ void init_buffers(float* __restrict__ m, float* __restrict__ norm,
                             float* __restrict__ agg, int N)
{
    int i = blockIdx.x * blockDim.x + threadIdx.x;
    if (i < N * NV) { m[i] = -INFINITY; norm[i] = 1e-12f; }
    if (i < N * DVDIM) agg[i] = 0.0f;
}

__global__ void edge_score(const int* __restrict__ ei,
                           const float* __restrict__ t, const float* __restrict__ s,
                           float* __restrict__ ebuf, float* __restrict__ m,
                           int E, int N)
{
    int tid = blockIdx.x * blockDim.x + threadIdx.x;
    int total = (E + N) * NV;
    if (tid >= total) return;
    int edge = tid / NV, v = tid - edge * NV;
    int src, dst;
    if (edge < E) { src = ei[edge]; dst = ei[E + edge]; }
    else          { src = dst = edge - E; }               // self loop
    const float* tp = t + (size_t)dst * PROJ + v * DVDIM;
    const float* sp = s + (size_t)src * PROJ + v * DVDIM;
    float acc = 0.0f;
#pragma unroll
    for (int j = 0; j < DVDIM; ++j) acc += tp[j] * sp[j];
    ebuf[tid] = acc;
    atomicMaxF(&m[(size_t)dst * NV + v], acc);
}

__global__ void edge_exp(const int* __restrict__ ei, float* __restrict__ ebuf,
                         const float* __restrict__ m, float* __restrict__ norm,
                         int E, int N)
{
    int tid = blockIdx.x * blockDim.x + threadIdx.x;
    int total = (E + N) * NV;
    if (tid >= total) return;
    int edge = tid / NV, v = tid - edge * NV;
    int dst = (edge < E) ? ei[E + edge] : (edge - E);
    float ec = __expf(ebuf[tid] - m[(size_t)dst * NV + v]);
    ebuf[tid] = ec;
    atomicAdd(&norm[(size_t)dst * NV + v], ec);
}

__global__ void edge_agg(const int* __restrict__ ei, const float* __restrict__ g,
                         const float* __restrict__ ebuf, const float* __restrict__ norm,
                         float* __restrict__ agg, int E, int N)
{
    int tid = blockIdx.x * blockDim.x + threadIdx.x;
    int total = (E + N) * DVDIM;
    if (tid >= total) return;
    int edge = tid >> 4, j = tid & 15;
    int src, dst;
    if (edge < E) { src = ei[edge]; dst = ei[E + edge]; }
    else          { src = dst = edge - E; }
    float acc = 0.0f;
#pragma unroll
    for (int v = 0; v < NV; ++v) {
        float alpha = ebuf[(size_t)edge * NV + v] / norm[(size_t)dst * NV + v];
        acc += g[(size_t)src * PROJ + v * DVDIM + j] * alpha;
    }
    atomicAdd(&agg[(size_t)dst * DVDIM + j], acc * (1.0f / NV));
}

// ---------------------------------------------------------------------------
// Final: logits = feat[N,128] @ Wout[128,8] + bout; row softmax. One wave/node.
// ---------------------------------------------------------------------------
__global__ void __launch_bounds__(256) logits_softmax(
    const float* __restrict__ feat, const float* __restrict__ Wout,
    const float* __restrict__ bout, float* __restrict__ out, int N)
{
    int node = (blockIdx.x * blockDim.x + threadIdx.x) >> 5;
    int lane = threadIdx.x & 31;
    if (node >= N) return;                    // wave-uniform
    const float* f = feat + (size_t)node * 128;
    float v0 = f[lane], v1 = f[lane + 32], v2 = f[lane + 64], v3 = f[lane + 96];
    float lg[8];
#pragma unroll
    for (int a = 0; a < 8; ++a) {
        float acc = v0 * Wout[(size_t)lane * 8 + a]
                  + v1 * Wout[(size_t)(lane + 32) * 8 + a]
                  + v2 * Wout[(size_t)(lane + 64) * 8 + a]
                  + v3 * Wout[(size_t)(lane + 96) * 8 + a];
#pragma unroll
        for (int off = 16; off; off >>= 1) acc += __shfl_xor(acc, off, 32);
        lg[a] = acc + bout[a];
    }
    float mx = lg[0];
#pragma unroll
    for (int a = 1; a < 8; ++a) mx = fmaxf(mx, lg[a]);
    float sum = 0.0f;
#pragma unroll
    for (int a = 0; a < 8; ++a) { lg[a] = __expf(lg[a] - mx); sum += lg[a]; }
    float inv = 1.0f / sum;
#pragma unroll
    for (int a = 0; a < 8; ++a)
        if (lane == a) out[(size_t)node * 8 + a] = lg[a] * inv;
}

// ---------------------------------------------------------------------------
extern "C" void kernel_launch(void* const* d_in, const int* in_sizes, int n_in,
                              void* d_out, int out_size, void* d_ws, size_t ws_size,
                              hipStream_t stream) {
    const float* x    = (const float*)d_in[0];
    const int*   ei   = (const int*)d_in[1];    // edge_index per harness int convention
    const float* W0   = (const float*)d_in[2];
    const float* b0   = (const float*)d_in[3];
    const float* W1   = (const float*)d_in[4];
    const float* b1   = (const float*)d_in[5];
    const float* Wt   = (const float*)d_in[6];
    const float* bt   = (const float*)d_in[7];
    const float* Ws   = (const float*)d_in[8];
    const float* bs   = (const float*)d_in[9];
    const float* Wh   = (const float*)d_in[10];
    const float* bh   = (const float*)d_in[11];
    const float* Wo   = (const float*)d_in[12];
    const float* bo   = (const float*)d_in[13];
    const float* Wout = (const float*)d_in[14];
    const float* bout = (const float*)d_in[15];

    const int N  = in_sizes[0] / 128;
    const int E  = in_sizes[1] / 2;
    const int EP = E + N;

    // workspace layout (floats, all offsets multiples of 8 floats = 32B)
    float* ws = (float*)d_ws;
    size_t off = 0;
    auto alloc = [&](size_t nf) { float* p = ws + off; off += (nf + 7) & ~(size_t)7; return p; };
    float* h0   = alloc((size_t)N * 128);   // reused as final features
    float* h1   = alloc((size_t)N * 128);
    float* t    = alloc((size_t)N * PROJ);
    float* s    = alloc((size_t)N * PROJ);
    float* g    = alloc((size_t)N * PROJ);
    float* ebuf = alloc((size_t)EP * NV);
    float* m    = alloc((size_t)N * NV);
    float* norm = alloc((size_t)N * NV);
    float* agg  = alloc((size_t)N * DVDIM);
    // packed f16 weight buffers (sizes: (M/16)*ceil(K/32)*512 f16)
    _Float16* W0p = (_Float16*)alloc(8 * 4 * 512 / 2);   // 128x128
    _Float16* W1p = (_Float16*)alloc(8 * 4 * 512 / 2);
    _Float16* Wtp = (_Float16*)alloc(5 * 4 * 512 / 2);   // 128x80
    _Float16* Wsp = (_Float16*)alloc(5 * 4 * 512 / 2);
    _Float16* Whp = (_Float16*)alloc(5 * 4 * 512 / 2);
    _Float16* Wop = (_Float16*)alloc(8 * 1 * 512 / 2);   // 16x128 (K padded to 32)
    float* feat = h0;                                     // reuse

    const dim3 blk(256);
    const int rowBlocks = (N + 127) / 128;

    // pack weights into WMMA B layout (tiny, L2-resident)
    pack_w_f16<<<(8 * 4 * 512 + 255) / 256, blk, 0, stream>>>(W0,   W0p, 128, 128);
    pack_w_f16<<<(8 * 4 * 512 + 255) / 256, blk, 0, stream>>>(W1,   W1p, 128, 128);
    pack_w_f16<<<(5 * 4 * 512 + 255) / 256, blk, 0, stream>>>(Wt,   Wtp, 128, PROJ);
    pack_w_f16<<<(5 * 4 * 512 + 255) / 256, blk, 0, stream>>>(Ws,   Wsp, 128, PROJ);
    pack_w_f16<<<(5 * 4 * 512 + 255) / 256, blk, 0, stream>>>(Wh,   Whp, 128, PROJ);
    pack_w_f16<<<(8 * 1 * 512 + 255) / 256, blk, 0, stream>>>(Wo,   Wop, 16,  128);

    // node MLP
    wmma_gemm_kernel<128, true><<<dim3(8, rowBlocks), blk, 0, stream>>>(x,  W0p, b0, h0, N, 128);
    wmma_gemm_kernel<128, true><<<dim3(8, rowBlocks), blk, 0, stream>>>(h0, W1p, b1, h1, N, 128);
    // attention projections hoisted per-node (ReLU commutes with gather)
    wmma_gemm_kernel<128, true><<<dim3(5, rowBlocks), blk, 0, stream>>>(h1, Wtp, bt, t, N, PROJ);
    wmma_gemm_kernel<128, true><<<dim3(5, rowBlocks), blk, 0, stream>>>(h1, Wsp, bs, s, N, PROJ);
    wmma_gemm_kernel<128, true><<<dim3(5, rowBlocks), blk, 0, stream>>>(h1, Whp, bh, g, N, PROJ);

    // segment softmax + aggregation over edges (+ self loops)
    init_buffers<<<(N * DVDIM + 255) / 256, blk, 0, stream>>>(m, norm, agg, N);
    edge_score<<<(EP * NV + 255) / 256, blk, 0, stream>>>(ei, t, s, ebuf, m, E, N);
    edge_exp  <<<(EP * NV + 255) / 256, blk, 0, stream>>>(ei, ebuf, m, norm, E, N);
    edge_agg  <<<(EP * DVDIM + 255) / 256, blk, 0, stream>>>(ei, g, ebuf, norm, agg, E, N);

    // output head
    wmma_gemm_kernel<16, true><<<dim3(8, rowBlocks), blk, 0, stream>>>(agg, Wop, bo, feat, N, 128);
    logits_softmax<<<(N * 32 + 255) / 256, blk, 0, stream>>>(feat, Wout, bout, (float*)d_out, N);
}
